// EncoderTARDIS_19052474925183
// MI455X (gfx1250) — matrix operand order repaired and checked by reading, not compiled, not run on hardware
//
#include <hip/hip_runtime.h>
#include <hip/hip_bf16.h>
#include <stdint.h>
#include <stddef.h>

// ---------------- problem dims ----------------
#define IDIM  512
#define HDIM  1024
#define NSLOT 128        // N memory slots
#define AKEY  64         // A
#define CVAL  64         // C
#define ACD   128        // A + C
#define TSTEP 256        // T
#define BSZ   128
#define KTOT  1664       // HDIM + IDIM + ACD (concatenated K for the cell GEMM)

typedef __bf16 bf16_t;
typedef __attribute__((ext_vector_type(16))) __bf16 bf16x16;
typedef __attribute__((ext_vector_type(8)))  __bf16 bf16x8;
typedef __attribute__((ext_vector_type(8)))  float  f32x8;

// ---------------- WMMA helpers (CDNA5 16x16x32 bf16 -> f32) ----------------
__device__ __forceinline__ f32x8 wmma_bf16(bf16x16 a, bf16x16 b, f32x8 c) {
    // (neg_a, A, neg_b, B, c_mod, C, reuse_a, reuse_b)
    return __builtin_amdgcn_wmma_f32_16x16x32_bf16(false, a, false, b, (short)0, c, false, false);
}

// A fragment, 16x32 bf16, source row-major with K contiguous (stride in elements).
// ISA layout: lane l holds row M=l%16; VGPR0..3 = K kh..kh+7, VGPR4..7 = K kh+16..kh+23, kh=(l/16)*8.
__device__ __forceinline__ bf16x16 load_a_frag(const bf16_t* base, int stride, int m0, int k0, int lane) {
    int row = m0 + (lane & 15);
    int kh  = (lane >> 4) * 8;
    const bf16_t* p = base + (size_t)row * stride + k0 + kh;
    bf16x8 lo = *reinterpret_cast<const bf16x8*>(p);       // 16B load
    bf16x8 hi = *reinterpret_cast<const bf16x8*>(p + 16);  // 16B load
    bf16x16 f;
#pragma unroll
    for (int i = 0; i < 8; ++i) { f[i] = lo[i]; f[i + 8] = hi[i]; }
    return f;
}

// B fragment, 32x16 bf16, source stored N-major (src[n][k], stride in elements).
// ISA layout: lane l holds col N=l%16; VGPR0..3 = K kh..kh+7, VGPR4..7 = K kh+16..kh+23, kh=(l/16)*8.
__device__ __forceinline__ bf16x16 load_b_frag(const bf16_t* base, int stride, int n0, int k0, int lane) {
    int col = n0 + (lane & 15);
    int kk  = k0 + (lane >> 4) * 8;
    const bf16_t* p = base + (size_t)col * stride + kk;
    bf16x8 lo = *reinterpret_cast<const bf16x8*>(p);
    bf16x8 hi = *reinterpret_cast<const bf16x8*>(p + 16);
    bf16x16 f;
#pragma unroll
    for (int i = 0; i < 8; ++i) { f[i] = lo[i]; f[i + 8] = hi[i]; }
    return f;
}

// ---------------- hash PRNG -> gumbel ----------------
__device__ __forceinline__ unsigned hash_u32(unsigned x) {
    x ^= x >> 16; x *= 0x7feb352dU;
    x ^= x >> 15; x *= 0x846ca68bU;
    x ^= x >> 16;
    return x;
}
__device__ __forceinline__ float gumbelf(unsigned key) {
    float u = ((hash_u32(key) >> 8) + 0.5f) * (1.0f / 16777216.0f);
    return -__logf(-__logf(u));
}
__device__ __forceinline__ float sigmoidf(float x) { return 1.0f / (1.0f + __expf(-x)); }

// ---------------- init kernels ----------------
__global__ void init_state(const float* __restrict__ mem_bias,
                           float* __restrict__ mem, float* __restrict__ wsum,
                           float* __restrict__ h0) {
    int b = blockIdx.x;
    for (int i = threadIdx.x; i < NSLOT * ACD; i += blockDim.x) mem[(size_t)b * NSLOT * ACD + i] = mem_bias[i];
    for (int i = threadIdx.x; i < NSLOT; i += blockDim.x)       wsum[b * NSLOT + i] = 0.0f;
    for (int i = threadIdx.x; i < HDIM; i += blockDim.x)        h0[b * HDIM + i] = 0.0f;
}

// Convert + transpose weights to N-major bf16 (done once per launch; L2-resident afterwards).
__global__ void conv_weights(const float* __restrict__ W_h2c, const float* __restrict__ W_i2c,
                             const float* __restrict__ W_r2c, const float* __restrict__ W_m2w,
                             const float* __restrict__ W_h2m,
                             bf16_t* __restrict__ Bct,   // [HDIM][KTOT]
                             bf16_t* __restrict__ Wmt,   // [ACD][ACD]
                             bf16_t* __restrict__ Whmt)  // [CVAL][HDIM]
{
    int idx = blockIdx.x * blockDim.x + threadIdx.x;
    int gsz = gridDim.x * blockDim.x;
    for (int i = idx; i < HDIM * KTOT; i += gsz) {
        int n = i / KTOT, k = i % KTOT;
        float v = (k < HDIM) ? W_h2c[(size_t)k * HDIM + n]
                : (k < HDIM + IDIM) ? W_i2c[(size_t)(k - HDIM) * HDIM + n]
                : W_r2c[(size_t)(k - HDIM - IDIM) * HDIM + n];
        Bct[i] = (bf16_t)v;
    }
    for (int i = idx; i < ACD * ACD; i += gsz) {
        int n = i >> 7, k = i & 127;
        Wmt[i] = (bf16_t)W_m2w[k * ACD + n];
    }
    for (int i = idx; i < CVAL * HDIM; i += gsz) {
        int n = i / HDIM, k = i % HDIM;
        Whmt[i] = (bf16_t)W_h2m[k * CVAL + n];
    }
}

// ---------------- K1: address / read / gates (one workgroup per batch) ----------------
#define MEMB_STRIDE 136   // 128 + 8 bf16 pad -> conflict-free 16B LDS loads
#define CL_STRIDE   132   // 128 + 4 f32 pad
#define SH_MEMB  (NSLOT * MEMB_STRIDE * 2)                    // 34816 B
#define SH_CL    (NSLOT * CL_STRIDE * 4)                      // 67584 B
#define SH_MISC  (544 * 4)                                    // u,s,part,red
#define SH_K1    (SH_MEMB + SH_CL + SH_MISC)

__global__ __launch_bounds__(256) void k1_addr(
    int t,
    const float* __restrict__ embs, const float* __restrict__ hprev,
    float* __restrict__ mem, float* __restrict__ wsum,
    const bf16_t* __restrict__ Wmt, const float* __restrict__ atten_base,
    const float* __restrict__ W_h2w, const float* __restrict__ W_i2w, const float* __restrict__ W_u2w,
    const float* __restrict__ W_h2g, const float* __restrict__ W_i2g, const float* __restrict__ W_r2g,
    const float* __restrict__ W_h2ab, const float* __restrict__ W_i2ab, const float* __restrict__ W_r2ab,
    bf16_t* __restrict__ Astep, float* __restrict__ fi_out, float* __restrict__ o_out,
    int* __restrict__ pos_out)
{
    extern __shared__ char smem_raw[];
    bf16_t* membf = (bf16_t*)smem_raw;                       // [NSLOT][MEMB_STRIDE]
    float*  Cl    = (float*)(smem_raw + SH_MEMB);            // [NSLOT][CL_STRIDE]
    float*  misc  = (float*)(smem_raw + SH_MEMB + SH_CL);
    float*  uS    = misc;          // 128: layernorm(w_sum), later reused for r
    float*  sS    = misc + 128;    // 128: broadcast pre-activation
    float*  part  = misc + 256;    // 256
    float*  redS  = misc + 512;    // 16
    int*    ipos  = (int*)(misc + 528);

    const int b    = blockIdx.x;
    const int tid  = threadIdx.x;
    const int lane = tid & 31;
    const int wave = tid >> 5;

    const float* memb = mem + (size_t)b * NSLOT * ACD;
    const float* emb  = embs + (size_t)t * BSZ * IDIM + (size_t)b * IDIM;
    const float* h    = hprev + (size_t)b * HDIM;
    float*       ws   = wsum + b * NSLOT;

    // stage mem[b] as bf16 into LDS (A matrix of the slot GEMM)
    for (int i = tid; i < NSLOT * ACD; i += 256) {
        int row = i >> 7, col = i & 127;
        membf[row * MEMB_STRIDE + col] = (bf16_t)memb[i];
    }
    // u = layer_norm(w_sum[b])
    if (tid < 128) part[tid] = ws[tid];
    __syncthreads();
    if (tid == 0) {
        float s = 0.f, ss = 0.f;
        for (int i = 0; i < NSLOT; ++i) { float v = part[i]; s += v; ss += v * v; }
        float mu = s * (1.0f / NSLOT);
        redS[0] = mu;
        redS[1] = rsqrtf(fmaxf(ss * (1.0f / NSLOT) - mu * mu, 0.0f) + 1e-5f);
    }
    __syncthreads();
    if (tid < 128) uS[tid] = (ws[tid] - redS[0]) * redS[1];
    __syncthreads();

    // s[j] = h@W_h2w + emb@W_i2w + u@W_u2w  (coalesced weight-column reads)
    if (tid < 128) {
        float acc = 0.f;
        for (int k = 0; k < HDIM; ++k)  acc += h[k]   * W_h2w[k * ACD + tid];
        for (int k = 0; k < IDIM; ++k)  acc += emb[k] * W_i2w[k * ACD + tid];
        for (int k = 0; k < NSLOT; ++k) acc += uS[k]  * W_u2w[k * ACD + tid];
        sS[tid] = acc;
    }
    __syncthreads();

    // WMMA: C[slot m][ac n] = mem_bf16 @ W_m2w   (128x128, K=128)
    {
        int m0 = wave * 16;
        f32x8 acc[8];
#pragma unroll
        for (int j = 0; j < 8; ++j) acc[j] = (f32x8){0, 0, 0, 0, 0, 0, 0, 0};
#pragma unroll
        for (int kk = 0; kk < 4; ++kk) {
            bf16x16 a = load_a_frag(membf, MEMB_STRIDE, m0, kk * 32, lane);
#pragma unroll
            for (int j = 0; j < 8; ++j) {
                bf16x16 bb = load_b_frag(Wmt, ACD, j * 16, kk * 32, lane);
                acc[j] = wmma_bf16(a, bb, acc[j]);
            }
        }
        int mrow = m0 + 8 * (lane >> 4);
#pragma unroll
        for (int j = 0; j < 8; ++j) {
            int ncol = j * 16 + (lane & 15);
#pragma unroll
            for (int v = 0; v < 8; ++v) Cl[(mrow + v) * CL_STRIDE + ncol] = acc[j][v];
        }
    }
    __syncthreads();

    // logits[n] = sum_k tanh(s[k] + C[n][k]) * atten_base[k]; forward of straight-through
    // softmax is the hard one-hot => only argmax(logits + gumbel) matters (tau drops out).
    {
        int n = tid & 127, half = tid >> 7;
        float acc = 0.f;
        for (int k = half * 64; k < half * 64 + 64; ++k)
            acc += tanhf(sS[k] + Cl[n * CL_STRIDE + k]) * atten_base[k];
        part[tid] = acc;
    }
    __syncthreads();
    if (tid < 128) {
        unsigned key = 0x9E3779B9u ^ (unsigned)((t * BSZ + b) * NSLOT + tid);
        part[tid] = part[tid] + part[tid + 128] + gumbelf(key);
    }
    __syncthreads();
    if (tid == 0) {
        int best = 0; float bv = part[0];
        for (int i = 1; i < NSLOT; ++i) if (part[i] > bv) { bv = part[i]; best = i; }
        *ipos = best;
        pos_out[b] = best;
        ws[best] += 1.0f;                    // w_sum += hard one-hot
    }
    __syncthreads();
    int amax = *ipos;

    // r = mem[b, amax, :]  (reuse uS)
    if (tid < 128) uS[tid] = memb[amax * ACD + tid];
    __syncthreads();

    // gates (f,i,o) and ab logits: 5 dot products of length KTOT over [h, emb, r]
    for (int c = 0; c < 5; ++c) {
        float a = 0.f;
        for (int k = tid; k < HDIM; k += 256) a += h[k]   * (c < 3 ? W_h2g[k * 3 + c] : W_h2ab[k * 2 + (c - 3)]);
        for (int k = tid; k < IDIM; k += 256) a += emb[k] * (c < 3 ? W_i2g[k * 3 + c] : W_i2ab[k * 2 + (c - 3)]);
        if (tid < 128)                        a += uS[tid] * (c < 3 ? W_r2g[tid * 3 + c] : W_r2ab[tid * 2 + (c - 3)]);
        part[tid] = a;
        __syncthreads();
        for (int st = 128; st >= 1; st >>= 1) {
            if (tid < st) part[tid] += part[tid + st];
            __syncthreads();
        }
        if (tid == 0) redS[2 + c] = part[0];
        __syncthreads();
    }
    if (tid == 0) {
        float f = sigmoidf(redS[2]), ig = sigmoidf(redS[3]), o = sigmoidf(redS[4]);
        fi_out[b] = f + ig;
        o_out[b]  = o;
        unsigned kb = 0xABCD1234u ^ (unsigned)(t * 4 * BSZ + b);
        float g0 = gumbelf(kb), g1 = gumbelf(kb + BSZ), g2 = gumbelf(kb + 2 * BSZ), g3 = gumbelf(kb + 3 * BSZ);
        // forward of hard gumbel-sigmoid: threshold at 0
        redS[8] = (redS[5] + g0 - g1 > 0.f) ? 1.0f : 0.0f;   // alpha
        redS[9] = (redS[6] + g2 - g3 > 0.f) ? 1.0f : 0.0f;   // beta
    }
    __syncthreads();
    float alpha = redS[8], beta = redS[9];

    // Build A row for cell GEMM: [beta*h | emb | alpha*r], bf16
    bf16_t* Ab = Astep + (size_t)b * KTOT;
    for (int k = tid; k < HDIM; k += 256) Ab[k] = (bf16_t)(beta * h[k]);
    for (int k = tid; k < IDIM; k += 256) Ab[HDIM + k] = (bf16_t)emb[k];
    if (tid < 128) Ab[HDIM + IDIM + tid] = (bf16_t)(alpha * uS[tid]);
}

// ---------------- K2: cell GEMM 128x1024, K=1664; fused h_n epilogue ----------------
__global__ __launch_bounds__(256) void k2_cell(
    int t,
    const bf16_t* __restrict__ Astep, const bf16_t* __restrict__ Bct,
    const float* __restrict__ fi, const float* __restrict__ og,
    float* __restrict__ out_all, bf16_t* __restrict__ hbf)
{
    const int lane = threadIdx.x & 31, wave = threadIdx.x >> 5;
    const int m0   = wave * 16;                 // batch rows
    const int n0   = blockIdx.x * 16;           // HDIM cols
    f32x8 acc = (f32x8){0, 0, 0, 0, 0, 0, 0, 0};
    for (int ks = 0; ks < KTOT / 32; ++ks) {
        int k0 = ks * 32;
        __builtin_prefetch(Bct + (size_t)(n0 + (lane & 15)) * KTOT + k0 + 64, 0, 0);
        bf16x16 a  = load_a_frag(Astep, KTOT, m0, k0, lane);
        bf16x16 bb = load_b_frag(Bct,   KTOT, n0, k0, lane);
        acc = wmma_bf16(a, bb, acc);
    }
    float* outT = out_all + (size_t)t * BSZ * HDIM;
    int mrow = m0 + 8 * (lane >> 4);
    int ncol = n0 + (lane & 15);
#pragma unroll
    for (int v = 0; v < 8; ++v) {
        int m = mrow + v;
        // h_n = o * tanh((f + i) * tanh(cpre_acc))
        float hn = og[m] * tanhf(fi[m] * tanhf(acc[v]));
        outT[(size_t)m * HDIM + ncol] = hn;
        hbf[(size_t)m * HDIM + ncol]  = (bf16_t)hn;
    }
}

// ---------------- K3: val = h_n @ W_h2m + b; scatter into mem[b, pos, A:] ----------------
__global__ __launch_bounds__(256) void k3_val(
    int t,
    const bf16_t* __restrict__ hbf, const bf16_t* __restrict__ Whmt,
    const float* __restrict__ b_h2m, const int* __restrict__ posArg,
    float* __restrict__ mem)
{
    const int lane = threadIdx.x & 31, wave = threadIdx.x >> 5;
    const int tile = blockIdx.x * 8 + wave;     // 32 tiles = 8 row-blocks x 4 col-blocks
    const int m0 = (tile & 7) * 16;
    const int n0 = (tile >> 3) * 16;
    f32x8 acc = (f32x8){0, 0, 0, 0, 0, 0, 0, 0};
    for (int ks = 0; ks < HDIM / 32; ++ks) {
        bf16x16 a  = load_a_frag(hbf,  HDIM, m0, ks * 32, lane);
        bf16x16 bb = load_b_frag(Whmt, HDIM, n0, ks * 32, lane);
        acc = wmma_bf16(a, bb, acc);
    }
    int mrow = m0 + 8 * (lane >> 4);
    int ncol = n0 + (lane & 15);
    float bias = b_h2m[ncol];
#pragma unroll
    for (int v = 0; v < 8; ++v) {
        int m = mrow + v;
        int pos = (t < NSLOT) ? t : posArg[m];
        mem[(size_t)m * NSLOT * ACD + pos * ACD + AKEY + ncol] = acc[v] + bias;
    }
}

// ---------------- K4: snapshot mem_res / ws_res at t == lens-1 ----------------
__global__ void k4_snapshot(int t, const int* __restrict__ lens,
                            const float* __restrict__ mem, const float* __restrict__ wsum,
                            float* __restrict__ mem_res, float* __restrict__ ws_res)
{
    int b = blockIdx.x;
    if (t != lens[b] - 1) return;
    for (int i = threadIdx.x; i < NSLOT * ACD; i += blockDim.x)
        mem_res[(size_t)b * NSLOT * ACD + i] = mem[(size_t)b * NSLOT * ACD + i];
    for (int i = threadIdx.x; i < NSLOT; i += blockDim.x)
        ws_res[b * NSLOT + i] = wsum[b * NSLOT + i];
}

// ---------------- final: hid = output[lens-1, b, :] ----------------
__global__ void gather_hid(const int* __restrict__ lens, const float* __restrict__ out_all,
                           float* __restrict__ hid)
{
    int b = blockIdx.x;
    const float* src = out_all + (size_t)(lens[b] - 1) * BSZ * HDIM + (size_t)b * HDIM;
    for (int i = threadIdx.x; i < HDIM; i += blockDim.x) hid[(size_t)b * HDIM + i] = src[i];
}

// ---------------- host-side orchestration ----------------
extern "C" void kernel_launch(void* const* d_in, const int* in_sizes, int n_in,
                              void* d_out, int out_size, void* d_ws, size_t ws_size,
                              hipStream_t stream) {
    (void)in_sizes; (void)n_in; (void)out_size; (void)ws_size;
    const float* embs       = (const float*)d_in[0];
    const int*   lens       = (const int*)  d_in[1];
    const float* mem_bias   = (const float*)d_in[2];
    const float* atten_base = (const float*)d_in[3];
    const float* W_h2w      = (const float*)d_in[4];
    const float* W_i2w      = (const float*)d_in[5];
    const float* W_m2w      = (const float*)d_in[6];
    const float* W_u2w      = (const float*)d_in[7];
    const float* W_h2g      = (const float*)d_in[8];
    const float* W_i2g      = (const float*)d_in[9];
    const float* W_r2g      = (const float*)d_in[10];
    const float* W_h2ab     = (const float*)d_in[11];
    const float* W_i2ab     = (const float*)d_in[12];
    const float* W_r2ab     = (const float*)d_in[13];
    const float* W_h2c      = (const float*)d_in[14];
    const float* W_i2c      = (const float*)d_in[15];
    const float* W_r2c      = (const float*)d_in[16];
    // d_in[17]=W_h2tau, d_in[18]=b_h2tau: forward output is independent of tau (straight-through
    // hard one-hot); intentionally unused.
    const float* W_h2m      = (const float*)d_in[19];
    const float* b_h2m      = (const float*)d_in[20];

    // outputs: output (T,B,H) | hid (1,B,H) | mem_res (B,N,AC) | ws_res (B,N)
    float* out_all = (float*)d_out;
    float* hid     = out_all + (size_t)TSTEP * BSZ * HDIM;
    float* mem_res = hid + (size_t)BSZ * HDIM;
    float* ws_res  = mem_res + (size_t)BSZ * NSLOT * ACD;

    // workspace carve (bytes)
    char* ws = (char*)d_ws;
    bf16_t* Bct   = (bf16_t*)(ws);                          // 1024*1664 bf16
    bf16_t* Wmt   = (bf16_t*)(ws + 3407872);                // 128*128 bf16
    bf16_t* Whmt  = (bf16_t*)(ws + 3440640);                // 64*1024 bf16
    float*  mem   = (float*) (ws + 3571712);                // 128*128*128 f32
    float*  wsum  = (float*) (ws + 11960320);               // 128*128 f32
    float*  h0    = (float*) (ws + 12025856);               // 128*1024 f32
    bf16_t* hbf   = (bf16_t*)(ws + 12550144);               // 128*1024 bf16
    bf16_t* Astep = (bf16_t*)(ws + 12812288);               // 128*1664 bf16
    float*  fi    = (float*) (ws + 13238272);               // 128
    float*  og    = (float*) (ws + 13238784);               // 128
    int*    pos   = (int*)   (ws + 13239296);               // 128

    init_state<<<BSZ, 256, 0, stream>>>(mem_bias, mem, wsum, h0);
    conv_weights<<<2048, 256, 0, stream>>>(W_h2c, W_i2c, W_r2c, W_m2w, W_h2m, Bct, Wmt, Whmt);

    for (int t = 0; t < TSTEP; ++t) {
        const float* hprev = (t == 0) ? h0 : out_all + (size_t)(t - 1) * BSZ * HDIM;
        k1_addr<<<BSZ, 256, SH_K1, stream>>>(
            t, embs, hprev, mem, wsum, Wmt, atten_base,
            W_h2w, W_i2w, W_u2w, W_h2g, W_i2g, W_r2g, W_h2ab, W_i2ab, W_r2ab,
            Astep, fi, og, pos);
        k2_cell<<<HDIM / 16, 256, 0, stream>>>(t, Astep, Bct, fi, og, out_all, hbf);
        k3_val<<<4, 256, 0, stream>>>(t, hbf, Whmt, b_h2m, pos, mem);
        k4_snapshot<<<BSZ, 256, 0, stream>>>(t, lens, mem, wsum, mem_res, ws_res);
    }
    gather_hid<<<BSZ, 256, 0, stream>>>(lens, out_all, hid);
}
